// Multishape_OffsetAttention_Decoder_90048284327985
// MI455X (gfx1250) — matrix operand order repaired and checked by reading, not compiled, not run on hardware
//
#include <hip/hip_runtime.h>
#include <hip/hip_bf16.h>

// MI455X / gfx1250, wave32. All GEMMs via V_WMMA_F32_16X16X32_F16 with
// double-buffered LDS staging: straight copies use the CDNA5 async DMA path
// (global_load_async_to_lds_b128, ASYNCcnt), transposed staging uses wide
// global loads + ds scatter. Per-batch attention tiles stay L2-resident
// (192MB L2). Workspace requirement: ~215 MB.

typedef _Float16 h16;
typedef __attribute__((ext_vector_type(8)))  _Float16 v8h;
typedef __attribute__((ext_vector_type(16))) _Float16 v16h;
typedef __attribute__((ext_vector_type(8)))  float    v8f;

#define BB 16
#define NN 2048
#define FF 1024
#define CC 256
#define HD 64
#define EPS_BN 1e-5f
#define LDP 56   // LDS row pitch in f16: 112B = 16B-aligned, 28-dword stride
                 // -> 16 distinct banks across a 16-lane group (conflict-free)

// ---------------------------------------------------------------------------
// Strided-batched GEMM, f16 in / f32 accumulate:
//   C[m,n] = sum_k A(m,k)*B[k*ldB+n] (+bias[m])
//   A(m,k) = aTrans ? A[k*ldA+m] : A[m*ldA+k]
// Block = 256 threads = 8 waves (2 x 4); block tile 128x128, K-step 32,
// ping-pong LDS buffers (one barrier per K-step: staging of step i+1 overlaps
// WMMA of step i). Each wave: 64x32 tile = 8 WMMAs/K-step.
// Requirements: Ncols%128==0, K%64==0 is not needed (K%32==0), M%64==0
// (M%128==0 when aTrans). Out: f32 Cf and/or f16 Ch.
// ---------------------------------------------------------------------------
__global__ __launch_bounds__(256)
void wmma_gemm(const h16* __restrict__ A, const h16* __restrict__ Bm,
               float* __restrict__ Cf, h16* __restrict__ Ch,
               const float* __restrict__ bias,
               int M, int Ncols, int K, int aTrans,
               long ldA, long batchA, long ldB, long batchB,
               long batchCf, long batchCh)
{
  __shared__ h16 As[2][128 * LDP];   // As[buf][row][k], row = m in block tile
  __shared__ h16 Bs[2][128 * LDP];   // Bs[buf][col][k], col = n in block tile

  const int tid   = threadIdx.x;
  const int lane  = tid & 31;
  const int wav   = tid >> 5;       // 0..7
  const int waveM = wav >> 2;       // 0..1 : 64 rows each
  const int waveN = wav & 3;        // 0..3 : 32 cols each
  const int m0blk = blockIdx.y * 128;
  const int n0blk = blockIdx.x * 128;
  const int m0w   = m0blk + waveM * 64;
  const int n0w   = n0blk + waveN * 32;
  const bool active = (m0w < M);    // wave-uniform; inactive waves still stage+barrier

  A  += (long)blockIdx.z * batchA;
  Bm += (long)blockIdx.z * batchB;

  const int lm = lane & 15;         // A: row-in-tile; B/C: col-in-tile
  const int lh = lane >> 4;

  v8f acc[4][2] = {};

  // ---- staging helpers -----------------------------------------------------
  auto stageA = [&](int buf, int k0) {
    if (!aTrans) {
      // straight copy: CDNA5 async DMA global->LDS, 16B per lane, ASYNCcnt
#pragma unroll
      for (int c = tid; c < 512; c += 256) {       // 128 rows x 4 chunks of 8 f16
        const int row  = c >> 2;
        const int koff = (c & 3) * 8;
        if (m0blk + row < M) {                     // OOB rows never read (inactive waves)
          const h16* gp = A + (long)(m0blk + row) * ldA + k0 + koff;
          const unsigned la =
              (unsigned)(unsigned long long)&As[buf][row * LDP + koff];
          asm volatile("global_load_async_to_lds_b128 %0, %1, off"
                       :: "v"(la), "v"((unsigned long long)gp) : "memory");
        }
      }
    } else {
      // A[k*ldA + m]: wide load along contiguous m, scatter-transpose into LDS
#pragma unroll
      for (int c = tid; c < 512; c += 256) {       // 32 k x 16 chunks of 8 f16
        const int kk   = c >> 4;
        const int moff = (c & 15) * 8;
        const v8h av = *(const v8h*)(A + (long)(k0 + kk) * ldA + m0blk + moff);
#pragma unroll
        for (int j = 0; j < 8; ++j) As[buf][(moff + j) * LDP + kk] = av[j];
      }
    }
  };
  auto stageB = [&](int buf, int k0) {
    // B row-major [K,N] -> Bs[col][k] (transposed for b128 fragment loads)
#pragma unroll
    for (int c = tid; c < 512; c += 256) {         // 32 k x 16 chunks of 8 f16
      const int kk   = c >> 4;
      const int noff = (c & 15) * 8;
      const v8h bv = *(const v8h*)(Bm + (long)(k0 + kk) * ldB + n0blk + noff);
#pragma unroll
      for (int j = 0; j < 8; ++j) Bs[buf][(noff + j) * LDP + kk] = bv[j];
    }
  };

  // ---- software-pipelined main loop (one barrier per K-step) ---------------
  stageA(0, 0);
  stageB(0, 0);
  if (!aTrans) asm volatile("s_wait_asynccnt 0x0" ::: "memory");
  __syncthreads();

  const int steps = K >> 5;
  for (int it = 0; it < steps; ++it) {
    const int cur = it & 1;
    if (it + 1 < steps) {          // issue next tile's staging (overlaps WMMAs)
      stageA(cur ^ 1, (it + 1) * 32);
      stageB(cur ^ 1, (it + 1) * 32);
    }
    if (active) {
      // B fragment: element e -> k = 8*lh + (e<8 ? e : e+8) : runs at +0, +16
      v16h bfr[2];
#pragma unroll
      for (int nt = 0; nt < 2; ++nt) {
        const h16* p = &Bs[cur][(waveN * 32 + nt * 16 + lm) * LDP + 8 * lh];
        union { v16h v; v8h h[2]; } u;
        u.h[0] = *(const v8h*)(p);
        u.h[1] = *(const v8h*)(p + 16);
        bfr[nt] = u.v;
      }
      // A fragment: element e -> k = 8*lh + (e<8 ? e : e+8) : runs at +0, +16
#pragma unroll
      for (int mt = 0; mt < 4; ++mt) {
        const h16* p = &As[cur][(waveM * 64 + mt * 16 + lm) * LDP + 8 * lh];
        union { v16h v; v8h h[2]; } u;
        u.h[0] = *(const v8h*)(p);
        u.h[1] = *(const v8h*)(p + 16);
        const v16h afr = u.v;
        acc[mt][0] = __builtin_amdgcn_wmma_f32_16x16x32_f16(false, afr, false, bfr[0], (short)0, acc[mt][0], false, false);
        acc[mt][1] = __builtin_amdgcn_wmma_f32_16x16x32_f16(false, afr, false, bfr[1], (short)0, acc[mt][1], false, false);
      }
    }
    if (!aTrans && it + 1 < steps)
      asm volatile("s_wait_asynccnt 0x0" ::: "memory");  // DMA done before barrier
    __syncthreads();
  }

  if (!active) return;
  // C/D 16x16 f32 layout: lane -> N=lane%16, VGPR r -> M = r + 8*(lane/16)
  const long zCf = (long)blockIdx.z * batchCf;
  const long zCh = (long)blockIdx.z * batchCh;
#pragma unroll
  for (int mt = 0; mt < 4; ++mt) {
#pragma unroll
    for (int r = 0; r < 8; ++r) {
      const int m = m0w + mt * 16 + r + 8 * lh;
      const float bv = bias ? bias[m] : 0.0f;
#pragma unroll
      for (int nt = 0; nt < 2; ++nt) {
        const float val = acc[mt][nt][r] + bv;
        const long off = (long)m * Ncols + n0w + nt * 16 + lm;
        if (Cf) Cf[zCf + off] = val;
        if (Ch) Ch[zCh + off] = (h16)val;
      }
    }
  }
}

// ---------------------------------------------------------------------------
__global__ void cvt_f32_to_f16(const float* __restrict__ s, h16* __restrict__ d, long n)
{
  const long i = (long)blockIdx.x * blockDim.x + threadIdx.x;
  if (i < n) d[i] = (h16)s[i];
}

// Encoder: base[b,o] = mlp_b[o] + mlp_w[o,0:F] . feature[b]
__global__ void mlp_base(const float* __restrict__ feature,
                         const float* __restrict__ mlp_w,
                         const float* __restrict__ mlp_b,
                         float* __restrict__ base)
{
  const int idx = blockIdx.x * blockDim.x + threadIdx.x;   // BB*CC
  const int b = idx / CC, o = idx % CC;
  const float* w = mlp_w + (long)o * (FF + 3);
  const float* f = feature + (long)b * FF;
  float s = mlp_b[o];
  for (int i = 0; i < FF; ++i) s += w[i] * f[i];
  base[idx] = s;
}

// x0h[b,c,n] = f16( base[b,c] + mlp_w[c,F:F+3] . prior[b,:,n] )
__global__ void mlp_prior(const float* __restrict__ prior,
                          const float* __restrict__ mlp_w,
                          const float* __restrict__ base,
                          h16* __restrict__ x0h)
{
  const long idx = (long)blockIdx.x * blockDim.x + threadIdx.x;  // BB*CC*NN
  const long n = idx % NN;
  const long c = (idx / NN) % CC;
  const long b = idx / ((long)NN * CC);
  const float* w  = mlp_w + c * (long)(FF + 3) + FF;
  const float* pr = prior + b * 3L * NN;
  x0h[idx] = (h16)(base[b * CC + c] + w[0] * pr[n] + w[1] * pr[NN + n] + w[2] * pr[2L * NN + n]);
}

// Row softmax: read f32 energies, write f16 probabilities
__global__ void softmax_row(const float* __restrict__ e, h16* __restrict__ o)
{
  const float* r = e + (long)blockIdx.x * NN;
  h16* w = o + (long)blockIdx.x * NN;
  __shared__ float red[256];
  const int t = threadIdx.x;

  float mx = -3.0e38f;
  for (int i = t; i < NN; i += 256) mx = fmaxf(mx, r[i]);
  red[t] = mx; __syncthreads();
  for (int st = 128; st > 0; st >>= 1) {
    if (t < st) red[t] = fmaxf(red[t], red[t + st]);
    __syncthreads();
  }
  mx = red[0]; __syncthreads();

  float s = 0.0f;
  for (int i = t; i < NN; i += 256) s += __expf(r[i] - mx);
  red[t] = s; __syncthreads();
  for (int st = 128; st > 0; st >>= 1) {
    if (t < st) red[t] += red[t + st];
    __syncthreads();
  }
  const float inv = 1.0f / red[0];
  for (int i = t; i < NN; i += 256) w[i] = (h16)(__expf(r[i] - mx) * inv);
}

// Column L1-renorm in place on f16 att: att[:,m] /= (1e-9 + sum_n att[n,m])
__global__ void col_renorm(h16* __restrict__ att)
{
  const int m = blockIdx.x * blockDim.x + threadIdx.x;
  float s = 0.0f;
  for (int n = 0; n < NN; ++n) s += (float)att[(long)n * NN + m];
  const float inv = 1.0f / (1e-9f + s);
  for (int n = 0; n < NN; ++n) {
    const long i = (long)n * NN + m;
    att[i] = (h16)((float)att[i] * inv);
  }
}

// u = f16( x - x_r )
__global__ void residual_sub(const h16* __restrict__ xh, long xBatch,
                             const float* __restrict__ xr, h16* __restrict__ u)
{
  const long idx = (long)blockIdx.x * blockDim.x + threadIdx.x;  // BB*CC*NN
  const long n = idx % NN;
  const long c = (idx / NN) % CC;
  const long b = idx / ((long)NN * CC);
  u[idx] = (h16)((float)xh[b * xBatch + c * (long)NN + n] - xr[idx]);
}

// BN statistics over (batch, length), biased variance. One block per channel.
__global__ void bn_stats(const float* __restrict__ y, long yBatch, int nbatch,
                         float* __restrict__ mean, float* __restrict__ var)
{
  const int c = blockIdx.x;
  const int t = threadIdx.x;
  __shared__ float s1[256], s2[256];
  float a = 0.0f, q = 0.0f;
  for (int b = 0; b < nbatch; ++b) {
    const float* p = y + (long)b * yBatch + (long)c * NN;
    for (int i = t; i < NN; i += 256) { const float v = p[i]; a += v; q += v * v; }
  }
  s1[t] = a; s2[t] = q; __syncthreads();
  for (int st = 128; st > 0; st >>= 1) {
    if (t < st) { s1[t] += s1[t + st]; s2[t] += s2[t + st]; }
    __syncthreads();
  }
  if (t == 0) {
    const float inv = 1.0f / (float)(nbatch * NN);
    const float m = s1[0] * inv;
    mean[c] = m;
    var[c] = s2[0] * inv - m * m;
  }
}

// out = (xh ? xh : 0) + leaky( g*(y-mean)*rsqrt(var+eps) + b ); dual f32/f16 out
__global__ void bn_leaky_residual(const float* __restrict__ y,
                                  const h16* __restrict__ xh, long xBatch,
                                  const float* __restrict__ g,
                                  const float* __restrict__ bb,
                                  const float* __restrict__ mean,
                                  const float* __restrict__ var,
                                  float* __restrict__ out32, long o32Batch,
                                  h16* __restrict__ out16, long o16Batch)
{
  const long idx = (long)blockIdx.x * blockDim.x + threadIdx.x;  // BB*CC*NN
  const long n = idx % NN;
  const long c = (idx / NN) % CC;
  const long b = idx / ((long)NN * CC);
  float t = g[c] * (y[idx] - mean[c]) * rsqrtf(var[c] + EPS_BN) + bb[c];
  t = t > 0.0f ? t : 0.2f * t;
  if (xh) t += (float)xh[b * xBatch + c * (long)NN + n];
  if (out32) out32[b * o32Batch + c * (long)NN + n] = t;
  if (out16) out16[b * o16Batch + c * (long)NN + n] = (h16)t;
}

// dc2 (M=3): y3[b,j,n] = dc2_b[j] + sum_c dc2_w[j,c]*h[b,c,n]
__global__ void dc2_conv(const float* __restrict__ h, const float* __restrict__ w,
                         const float* __restrict__ bias, float* __restrict__ y3)
{
  const long idx = (long)blockIdx.x * blockDim.x + threadIdx.x;  // BB*NN
  const long n = idx % NN;
  const long b = idx / NN;
  const float* H = h + b * (long)CC * NN + n;
  float a0 = bias[0], a1 = bias[1], a2 = bias[2];
  for (int c = 0; c < CC; ++c) {
    const float hv = H[(long)c * NN];
    a0 += w[c] * hv;
    a1 += w[CC + c] * hv;
    a2 += w[2 * CC + c] * hv;
  }
  float* o = y3 + b * 3L * NN;
  o[n] = a0; o[NN + n] = a1; o[2L * NN + n] = a2;
}

// out[b,n,j] = tanh( g*(y3[b,j,n]-mean)*rsqrt(var+eps) + b )
__global__ void final_tanh(const float* __restrict__ y3,
                           const float* __restrict__ g, const float* __restrict__ bb,
                           const float* __restrict__ mean, const float* __restrict__ var,
                           float* __restrict__ out)
{
  const long idx = (long)blockIdx.x * blockDim.x + threadIdx.x;  // BB*NN*3
  const int  j = (int)(idx % 3);
  const long n = (idx / 3) % NN;
  const long b = idx / (3L * NN);
  const float v = y3[b * 3L * NN + (long)j * NN + n];
  out[idx] = tanhf(g[j] * (v - mean[j]) * rsqrtf(var[j] + EPS_BN) + bb[j]);
}

// ---------------------------------------------------------------------------
extern "C" void kernel_launch(void* const* d_in, const int* in_sizes, int n_in,
                              void* d_out, int out_size, void* d_ws, size_t ws_size,
                              hipStream_t stream)
{
  const float* feature = (const float*)d_in[0];
  const float* prior   = (const float*)d_in[1];
  const float* mlp_w   = (const float*)d_in[2];
  const float* mlp_b   = (const float*)d_in[3];
  const float* qk_w    = (const float*)d_in[4];
  const float* qk_b    = (const float*)d_in[5];
  const float* v_w     = (const float*)d_in[6];
  const float* v_b     = (const float*)d_in[7];
  const float* t_w     = (const float*)d_in[8];
  const float* t_b     = (const float*)d_in[9];
  const float* bn_g    = (const float*)d_in[10];
  const float* bn_b    = (const float*)d_in[11];
  const float* dc1_w   = (const float*)d_in[12];
  const float* dc1_g   = (const float*)d_in[13];
  const float* dc1_b   = (const float*)d_in[14];
  const float* dc2_w   = (const float*)d_in[15];
  const float* dc2_bias= (const float*)d_in[16];
  const float* dc2_g   = (const float*)d_in[17];
  const float* dc2_bb  = (const float*)d_in[18];

  const long CN  = (long)CC * NN;
  const long C4N = 4L * CC * NN;

  // -------- workspace carve (~215 MB), f32 region then f16 region --------
  float* wsf = (float*)d_ws;
  size_t off = 0;
  float* att32 = wsf + off; off += (size_t)NN * NN;        // 16.8 MB
  float* xr    = wsf + off; off += (size_t)BB * CC * NN;   // 33.5 MB (reused as h)
  float* y     = wsf + off; off += (size_t)BB * CC * NN;   // 33.5 MB
  float* y3    = wsf + off; off += (size_t)BB * 3 * NN;
  float* base  = wsf + off; off += (size_t)BB * CC;
  float* bnm   = wsf + off; off += (size_t)CC;
  float* bnv   = wsf + off; off += (size_t)CC + 64;        // keep f16 region 16B aligned
  h16* wsh = (h16*)(wsf + off);
  size_t hoff = 0;
  h16* x0h   = wsh + hoff; hoff += (size_t)BB * CC * NN;     // 16.8 MB
  h16* pch   = wsh + hoff; hoff += (size_t)BB * 4 * CC * NN; // 67.1 MB
  h16* qkh   = wsh + hoff; hoff += (size_t)BB * HD * NN;     //  4.2 MB
  h16* vvh   = wsh + hoff; hoff += (size_t)BB * CC * NN;     // 16.8 MB
  h16* att16 = wsh + hoff; hoff += (size_t)NN * NN;          //  8.4 MB
  h16* uh    = wsh + hoff; hoff += (size_t)BB * CC * NN;     // 16.8 MB
  h16* qk_w16  = wsh + hoff; hoff += (size_t)4 * HD * CC;
  h16* v_w16   = wsh + hoff; hoff += (size_t)4 * CC * CC;
  h16* t_w16   = wsh + hoff; hoff += (size_t)4 * CC * CC;
  h16* dc1_w16 = wsh + hoff; hoff += (size_t)CC * 4 * CC;
  float* h = xr;   // xr dead after the attention layers

  const int ELT = (BB * CC * NN) / 256;

  // -------- weight conversion (tiny, once per call) --------
  cvt_f32_to_f16<<<(4 * HD * CC) / 256, 256, 0, stream>>>(qk_w, qk_w16, 4L * HD * CC);
  cvt_f32_to_f16<<<(4 * CC * CC) / 256, 256, 0, stream>>>(v_w, v_w16, 4L * CC * CC);
  cvt_f32_to_f16<<<(4 * CC * CC) / 256, 256, 0, stream>>>(t_w, t_w16, 4L * CC * CC);
  cvt_f32_to_f16<<<(4 * CC * CC) / 256, 256, 0, stream>>>(dc1_w, dc1_w16, 4L * CC * CC);

  // -------- encoder --------
  mlp_base <<<(BB * CC) / 256, 256, 0, stream>>>(feature, mlp_w, mlp_b, base);
  mlp_prior<<<ELT, 256, 0, stream>>>(prior, mlp_w, base, x0h);

  // -------- 4 offset-attention layers --------
  for (int i = 0; i < 4; ++i) {
    const h16* xh = (i == 0) ? x0h : pch + (size_t)(i - 1) * CC * NN;
    const long xBatch = (i == 0) ? CN : C4N;

    // q/k conv (shared weights) -> qkh[b, 0:64, 0:N]  (f16 out only)
    wmma_gemm<<<dim3(NN / 128, 1, BB), 256, 0, stream>>>(
        qk_w16 + (size_t)i * HD * CC, xh, nullptr, qkh, qk_b + i * HD,
        HD, NN, CC, 0,
        CC, 0, NN, xBatch,
        0, (long)HD * NN);

    // v conv -> vvh[b, 0:256, 0:N]  (f16 out only)
    wmma_gemm<<<dim3(NN / 128, CC / 128, BB), 256, 0, stream>>>(
        v_w16 + (size_t)i * CC * CC, xh, nullptr, vvh, v_b + i * CC,
        CC, NN, CC, 0,
        CC, 0, NN, xBatch,
        0, CN);

    // attention per batch: att tiles stay L2-resident
    for (int b = 0; b < BB; ++b) {
      const h16* qkb = qkh + (size_t)b * HD * NN;
      // energy = qk^T . qk   [N,N], K=64 (A transposed view)
      wmma_gemm<<<dim3(NN / 128, NN / 128, 1), 256, 0, stream>>>(
          qkb, qkb, att32, nullptr, nullptr,
          NN, NN, HD, 1,
          NN, 0, NN, 0,
          0, 0);
      softmax_row<<<NN, 256, 0, stream>>>(att32, att16);
      col_renorm <<<NN / 256, 256, 0, stream>>>(att16);
      // x_r = v . att  -> f32
      wmma_gemm<<<dim3(NN / 128, CC / 128, 1), 256, 0, stream>>>(
          vvh + (size_t)b * CN, att16, xr + (size_t)b * CN, nullptr, nullptr,
          CC, NN, NN, 0,
          NN, 0, NN, 0,
          0, 0);
    }

    // u = x - x_r  (f16)
    residual_sub<<<ELT, 256, 0, stream>>>(xh, xBatch, xr, uh);

    // t conv: y = t_w . u + t_b  (f32)
    wmma_gemm<<<dim3(NN / 128, CC / 128, BB), 256, 0, stream>>>(
        t_w16 + (size_t)i * CC * CC, uh, y, nullptr, t_b + i * CC,
        CC, NN, CC, 0,
        CC, 0, NN, CN,
        CN, 0);

    // BN (training-mode) + leaky + residual -> pch slice (f16)
    bn_stats<<<CC, 256, 0, stream>>>(y, CN, BB, bnm, bnv);
    bn_leaky_residual<<<ELT, 256, 0, stream>>>(
        y, xh, xBatch, bn_g + i * CC, bn_b + i * CC, bnm, bnv,
        nullptr, 0, pch + (size_t)i * CC * NN, C4N);
  }

  // -------- decoder --------
  // dc1: y = dc1_w [256,1024] . pc [1024,N]  (no bias)
  wmma_gemm<<<dim3(NN / 128, CC / 128, BB), 256, 0, stream>>>(
      dc1_w16, pch, y, nullptr, nullptr,
      CC, NN, 4 * CC, 0,
      4 * CC, 0, NN, C4N,
      CN, 0);
  bn_stats<<<CC, 256, 0, stream>>>(y, CN, BB, bnm, bnv);
  bn_leaky_residual<<<ELT, 256, 0, stream>>>(
      y, nullptr, 0, dc1_g, dc1_b, bnm, bnv, h, CN, nullptr, 0);

  // dc2 (M=3) plain, then BN + tanh into [B,N,3]
  dc2_conv<<<(BB * NN) / 256, 256, 0, stream>>>(h, dc2_w, dc2_bias, y3);
  bn_stats<<<3, 256, 0, stream>>>(y3, 3L * NN, BB, bnm, bnv);
  final_tanh<<<(BB * NN * 3) / 256, 256, 0, stream>>>(
      y3, dc2_g, dc2_bb, bnm, bnv, (float*)d_out);
}
